// dMoE_16535624089677
// MI455X (gfx1250) — matrix operand order repaired and checked by reading, not compiled, not run on hardware
//
#include <hip/hip_runtime.h>
#include <stdint.h>

#define T_TOK 4096
#define HID   1024
#define NEXP  8
#define FFN   4096
#define TOPK  2
#define LDA   40   // LDS row stride in bf16 units (80 B, 16-B aligned)

typedef __attribute__((ext_vector_type(16))) __bf16 v16bf;
typedef __attribute__((ext_vector_type(8)))  float  v8f;

union FragBF { v16bf v; uint4 u[2]; };

__device__ __forceinline__ uint32_t pk_bf16(float a, float b) {
  union { float f; uint32_t u; } ca, cb; ca.f = a; cb.f = b;
  uint32_t ua = ca.u + 0x7FFFu + ((ca.u >> 16) & 1u);
  uint32_t ub = cb.u + 0x7FFFu + ((cb.u >> 16) & 1u);
  return (ua >> 16) | (ub & 0xFFFF0000u);
}
__device__ __forceinline__ uint16_t f2bf(float a) {
  union { float f; uint32_t u; } c; c.f = a;
  return (uint16_t)((c.u + 0x7FFFu + ((c.u >> 16) & 1u)) >> 16);
}

// ---------------- Router: scores, softmax, top-2, per-expert counts ----------
__global__ __launch_bounds__(256) void moe_router(
    const float* __restrict__ x, const float* __restrict__ rw,
    int* __restrict__ counts, int* __restrict__ tokE, float* __restrict__ tokW) {
  __shared__ float s_rw[NEXP * HID];
  for (int i = threadIdx.x * 4; i < NEXP * HID; i += 256 * 4)
    *(float4*)&s_rw[i] = *(const float4*)&rw[i];
  __syncthreads();
  const int wave = threadIdx.x >> 5, lane = threadIdx.x & 31;
  const int t = blockIdx.x * 8 + wave;
  const float* xt = x + (size_t)t * HID;
  float acc[NEXP];
#pragma unroll
  for (int e = 0; e < NEXP; ++e) acc[e] = 0.f;
  for (int h = lane; h < HID; h += 32) {
    float xv = xt[h];
#pragma unroll
    for (int e = 0; e < NEXP; ++e) acc[e] += xv * s_rw[e * HID + h];
  }
#pragma unroll
  for (int e = 0; e < NEXP; ++e)
#pragma unroll
    for (int m = 16; m >= 1; m >>= 1) acc[e] += __shfl_xor(acc[e], m, 32);
  if (lane == 0) {
    float mx = acc[0];
#pragma unroll
    for (int e = 1; e < NEXP; ++e) mx = fmaxf(mx, acc[e]);
    float p[NEXP], s = 0.f;
#pragma unroll
    for (int e = 0; e < NEXP; ++e) { p[e] = __expf(acc[e] - mx); s += p[e]; }
    float inv = 1.f / s;
#pragma unroll
    for (int e = 0; e < NEXP; ++e) p[e] *= inv;
    int e0 = 0;
#pragma unroll
    for (int e = 1; e < NEXP; ++e) if (p[e] > p[e0]) e0 = e;
    int e1 = (e0 == 0) ? 1 : 0;
#pragma unroll
    for (int e = 0; e < NEXP; ++e) if (e != e0 && p[e] > p[e1]) e1 = e;
    tokE[2 * t + 0] = e0; tokW[2 * t + 0] = p[e0];
    tokE[2 * t + 1] = e1; tokW[2 * t + 1] = p[e1];
    atomicAdd(&counts[e0], 1);
    atomicAdd(&counts[e1], 1);
  }
}

// ---------------- Exclusive scan of 8 counts -------------------------------
__global__ void moe_scan(const int* __restrict__ counts, int* __restrict__ offs) {
  if (threadIdx.x == 0 && blockIdx.x == 0) {
    int a = 0;
    for (int e = 0; e < NEXP; ++e) { offs[e] = a; a += counts[e]; }
    offs[NEXP] = a;
  }
}

// ---------------- Build compacted per-expert row lists ---------------------
__global__ __launch_bounds__(256) void moe_assign(
    const int* __restrict__ tokE, const int* __restrict__ offs,
    int* __restrict__ cursor, int* __restrict__ rowTok, int* __restrict__ rowOf) {
  int t = blockIdx.x * 256 + threadIdx.x;
  if (t >= T_TOK) return;
#pragma unroll
  for (int k = 0; k < TOPK; ++k) {
    int e = tokE[2 * t + k];
    int slot = atomicAdd(&cursor[e], 1);
    int r = offs[e] + slot;
    rowTok[r] = t;
    rowOf[2 * t + k] = r;
  }
}

// ---------------- Stage 1: G = gelu(X@w1^T) * (X@v1^T), bf16 WMMA ----------
// Double-buffered LDS pipeline: global loads of tile k+1 issue before the
// WMMA compute of tile k, hiding HBM latency under the matrix pipe.
__global__ __launch_bounds__(256) void moe_glu(
    const float* __restrict__ x, const float* __restrict__ w1,
    const float* __restrict__ v1, const int* __restrict__ offs,
    const int* __restrict__ rowTok, uint16_t* __restrict__ G) {
  const int e = blockIdx.z;
  const int beg = offs[e], cnt = offs[e + 1] - beg;
  const int m0 = blockIdx.y * 128;
  if (m0 >= cnt) return;
  const int f0 = blockIdx.x * 64;

  __shared__ int s_tok[128];
  __shared__ __align__(16) uint16_t sA[2][128 * LDA];
  __shared__ __align__(16) uint16_t sB1[2][64 * LDA];
  __shared__ __align__(16) uint16_t sB2[2][64 * LDA];

  const int tid = threadIdx.x;
  if (tid < 128) {
    int r = m0 + tid; if (r > cnt - 1) r = cnt - 1;
    s_tok[tid] = rowTok[beg + r];
  }
  __syncthreads();

  const int wv = tid >> 5, lane = tid & 31;
  const int wm = wv & 1, wn = wv >> 1;      // 2x4 wave grid -> 64x16 per wave
  const int bRow = wn * 16 + (lane & 15);
  const int kb = (lane >> 4) * 16;
  const int kbase = (lane >> 4) * 16;

  v8f zero = {0.f, 0.f, 0.f, 0.f, 0.f, 0.f, 0.f, 0.f};
  v8f acc1[4], acc2[4];
#pragma unroll
  for (int i = 0; i < 4; ++i) { acc1[i] = zero; acc2[i] = zero; }

  float4 ax[4], b1x[2], b2x[2];

  // ---- prologue: fetch tile 0 and stage into buffer 0 ----
#pragma unroll
  for (int i = 0; i < 4; ++i) {
    int idx = tid + 256 * i, r = idx >> 3, c4 = (idx & 7) * 4;
    ax[i] = *(const float4*)(x + (size_t)s_tok[r] * HID + c4);
  }
#pragma unroll
  for (int i = 0; i < 2; ++i) {
    int idx = tid + 256 * i, r = idx >> 3, c4 = (idx & 7) * 4;
    size_t woff = ((size_t)e * FFN + f0 + r) * HID + c4;
    b1x[i] = *(const float4*)(w1 + woff);
    b2x[i] = *(const float4*)(v1 + woff);
  }
#pragma unroll
  for (int i = 0; i < 4; ++i) {
    int idx = tid + 256 * i, r = idx >> 3, c4 = (idx & 7) * 4;
    uint32_t* d = (uint32_t*)&sA[0][r * LDA + c4];
    d[0] = pk_bf16(ax[i].x, ax[i].y); d[1] = pk_bf16(ax[i].z, ax[i].w);
  }
#pragma unroll
  for (int i = 0; i < 2; ++i) {
    int idx = tid + 256 * i, r = idx >> 3, c4 = (idx & 7) * 4;
    uint32_t* d1 = (uint32_t*)&sB1[0][r * LDA + c4];
    d1[0] = pk_bf16(b1x[i].x, b1x[i].y); d1[1] = pk_bf16(b1x[i].z, b1x[i].w);
    uint32_t* d2 = (uint32_t*)&sB2[0][r * LDA + c4];
    d2[0] = pk_bf16(b2x[i].x, b2x[i].y); d2[1] = pk_bf16(b2x[i].z, b2x[i].w);
  }
  __syncthreads();

  const int NK = HID / 32;
  for (int it = 0; it < NK; ++it) {
    const int cur = it & 1, nxt = cur ^ 1;
    const bool more = (it + 1 < NK);
    const int kn = (it + 1) * 32;

    if (more) {  // issue next-tile global loads; latency hides under WMMA
#pragma unroll
      for (int i = 0; i < 4; ++i) {
        int idx = tid + 256 * i, r = idx >> 3, c4 = (idx & 7) * 4;
        ax[i] = *(const float4*)(x + (size_t)s_tok[r] * HID + kn + c4);
      }
#pragma unroll
      for (int i = 0; i < 2; ++i) {
        int idx = tid + 256 * i, r = idx >> 3, c4 = (idx & 7) * 4;
        size_t woff = ((size_t)e * FFN + f0 + r) * HID + kn + c4;
        __builtin_prefetch(w1 + woff + 32, 0, 1);
        b1x[i] = *(const float4*)(w1 + woff);
        b2x[i] = *(const float4*)(v1 + woff);
      }
    }

    // ---- compute from buffers[cur] ----
    {
      FragBF fb1, fb2;
      fb1.u[0] = *(const uint4*)&sB1[cur][bRow * LDA + kb];
      fb1.u[1] = *(const uint4*)&sB1[cur][bRow * LDA + kb + 8];
      fb2.u[0] = *(const uint4*)&sB2[cur][bRow * LDA + kb];
      fb2.u[1] = *(const uint4*)&sB2[cur][bRow * LDA + kb + 8];
#pragma unroll
      for (int mi = 0; mi < 4; ++mi) {
        FragBF fa;
        int aRow = wm * 64 + mi * 16 + (lane & 15);
        fa.u[0] = *(const uint4*)&sA[cur][aRow * LDA + kbase];
        fa.u[1] = *(const uint4*)&sA[cur][aRow * LDA + kbase + 8];
        acc1[mi] = __builtin_amdgcn_wmma_f32_16x16x32_bf16(
            false, fa.v, false, fb1.v, (short)0, acc1[mi], false, false);
        acc2[mi] = __builtin_amdgcn_wmma_f32_16x16x32_bf16(
            false, fa.v, false, fb2.v, (short)0, acc2[mi], false, false);
      }
    }

    if (more) {
      __syncthreads();  // everyone done reading buffers[nxt] from iter-1
#pragma unroll
      for (int i = 0; i < 4; ++i) {
        int idx = tid + 256 * i, r = idx >> 3, c4 = (idx & 7) * 4;
        uint32_t* d = (uint32_t*)&sA[nxt][r * LDA + c4];
        d[0] = pk_bf16(ax[i].x, ax[i].y); d[1] = pk_bf16(ax[i].z, ax[i].w);
      }
#pragma unroll
      for (int i = 0; i < 2; ++i) {
        int idx = tid + 256 * i, r = idx >> 3, c4 = (idx & 7) * 4;
        uint32_t* d1 = (uint32_t*)&sB1[nxt][r * LDA + c4];
        d1[0] = pk_bf16(b1x[i].x, b1x[i].y); d1[1] = pk_bf16(b1x[i].z, b1x[i].w);
        uint32_t* d2 = (uint32_t*)&sB2[nxt][r * LDA + c4];
        d2[0] = pk_bf16(b2x[i].x, b2x[i].y); d2[1] = pk_bf16(b2x[i].z, b2x[i].w);
      }
      __syncthreads();  // buffers[nxt] ready
    }
  }

  const int colg = f0 + wn * 16 + (lane & 15);
#pragma unroll
  for (int mi = 0; mi < 4; ++mi)
#pragma unroll
    for (int v = 0; v < 8; ++v) {
      int rloc = wm * 64 + mi * 16 + v + ((lane >> 4) << 3);
      if (m0 + rloc < cnt) {
        float a = acc1[mi][v];
        float g = 0.5f * a * (1.f + tanhf(0.7978845608028654f * (a + 0.044715f * a * a * a)));
        G[(size_t)(beg + m0 + rloc) * FFN + colg] = f2bf(g * acc2[mi][v]);
      }
    }
}

// ---------------- Stage 2: Y = G @ w2[e]  (K = F, bf16 WMMA) ---------------
__global__ __launch_bounds__(256) void moe_down(
    const uint16_t* __restrict__ G, const float* __restrict__ w2,
    const int* __restrict__ offs, float* __restrict__ Y) {
  const int e = blockIdx.z;
  const int beg = offs[e], cnt = offs[e + 1] - beg;
  const int m0 = blockIdx.y * 128;
  if (m0 >= cnt) return;
  const int h0 = blockIdx.x * 64;

  __shared__ __align__(16) uint16_t sA[2][128 * LDA];
  __shared__ __align__(16) uint16_t sBt[2][64 * LDA];

  const int tid = threadIdx.x;
  const int wv = tid >> 5, lane = tid & 31;
  const int wm = wv & 1, wn = wv >> 1;
  const int bRow = wn * 16 + (lane & 15);
  const int kb = (lane >> 4) * 16;
  const int kbase = (lane >> 4) * 16;

  v8f zero = {0.f, 0.f, 0.f, 0.f, 0.f, 0.f, 0.f, 0.f};
  v8f acc[4];
#pragma unroll
  for (int i = 0; i < 4; ++i) acc[i] = zero;

  uint4 gax[2]; float4 bx[2];

  // ---- prologue: tile 0 into buffer 0 ----
#pragma unroll
  for (int i = 0; i < 2; ++i) {
    int idx = tid + 256 * i, r = idx >> 2, c8 = (idx & 3) * 8;
    gax[i] = *(const uint4*)(G + (size_t)(beg + m0 + r) * FFN + c8);
  }
#pragma unroll
  for (int i = 0; i < 2; ++i) {
    int idx = tid + 256 * i, kk = idx >> 4, n4 = (idx & 15) * 4;
    bx[i] = *(const float4*)(w2 + ((size_t)e * FFN + kk) * HID + h0 + n4);
  }
#pragma unroll
  for (int i = 0; i < 2; ++i) {
    int idx = tid + 256 * i, r = idx >> 2, c8 = (idx & 3) * 8;
    *(uint4*)&sA[0][r * LDA + c8] = gax[i];
  }
#pragma unroll
  for (int i = 0; i < 2; ++i) {
    int idx = tid + 256 * i, kk = idx >> 4, n4 = (idx & 15) * 4;
    sBt[0][(n4 + 0) * LDA + kk] = f2bf(bx[i].x);
    sBt[0][(n4 + 1) * LDA + kk] = f2bf(bx[i].y);
    sBt[0][(n4 + 2) * LDA + kk] = f2bf(bx[i].z);
    sBt[0][(n4 + 3) * LDA + kk] = f2bf(bx[i].w);
  }
  __syncthreads();

  const int NK = FFN / 32;
  for (int it = 0; it < NK; ++it) {
    const int cur = it & 1, nxt = cur ^ 1;
    const bool more = (it + 1 < NK);
    const int kn = (it + 1) * 32;

    if (more) {
#pragma unroll
      for (int i = 0; i < 2; ++i) {
        int idx = tid + 256 * i, r = idx >> 2, c8 = (idx & 3) * 8;
        gax[i] = *(const uint4*)(G + (size_t)(beg + m0 + r) * FFN + kn + c8);
      }
#pragma unroll
      for (int i = 0; i < 2; ++i) {
        int idx = tid + 256 * i, kk = idx >> 4, n4 = (idx & 15) * 4;
        size_t woff = ((size_t)e * FFN + kn + kk) * HID + h0 + n4;
        __builtin_prefetch(w2 + woff + 32 * HID, 0, 1);
        bx[i] = *(const float4*)(w2 + woff);
      }
    }

    {
      FragBF fb;
      fb.u[0] = *(const uint4*)&sBt[cur][bRow * LDA + kb];
      fb.u[1] = *(const uint4*)&sBt[cur][bRow * LDA + kb + 8];
#pragma unroll
      for (int mi = 0; mi < 4; ++mi) {
        FragBF fa;
        int aRow = wm * 64 + mi * 16 + (lane & 15);
        fa.u[0] = *(const uint4*)&sA[cur][aRow * LDA + kbase];
        fa.u[1] = *(const uint4*)&sA[cur][aRow * LDA + kbase + 8];
        acc[mi] = __builtin_amdgcn_wmma_f32_16x16x32_bf16(
            false, fa.v, false, fb.v, (short)0, acc[mi], false, false);
      }
    }

    if (more) {
      __syncthreads();
#pragma unroll
      for (int i = 0; i < 2; ++i) {
        int idx = tid + 256 * i, r = idx >> 2, c8 = (idx & 3) * 8;
        *(uint4*)&sA[nxt][r * LDA + c8] = gax[i];
      }
#pragma unroll
      for (int i = 0; i < 2; ++i) {
        int idx = tid + 256 * i, kk = idx >> 4, n4 = (idx & 15) * 4;
        sBt[nxt][(n4 + 0) * LDA + kk] = f2bf(bx[i].x);
        sBt[nxt][(n4 + 1) * LDA + kk] = f2bf(bx[i].y);
        sBt[nxt][(n4 + 2) * LDA + kk] = f2bf(bx[i].z);
        sBt[nxt][(n4 + 3) * LDA + kk] = f2bf(bx[i].w);
      }
      __syncthreads();
    }
  }

  const int col = h0 + wn * 16 + (lane & 15);
#pragma unroll
  for (int mi = 0; mi < 4; ++mi)
#pragma unroll
    for (int v = 0; v < 8; ++v) {
      int rloc = wm * 64 + mi * 16 + v + ((lane >> 4) << 3);
      if (m0 + rloc < cnt)
        Y[(size_t)(beg + m0 + rloc) * HID + col] = acc[mi][v];
    }
}

// ---------------- Combine: out[t] = w0*Y[row0] + w1*Y[row1] ----------------
__global__ __launch_bounds__(256) void moe_combine(
    const float* __restrict__ Y, const int* __restrict__ rowOf,
    const float* __restrict__ tokW, float* __restrict__ out) {
  int i = blockIdx.x * 256 + threadIdx.x;       // float4 index
  int t = i >> 8;                               // HID/4 = 256 per token
  int c = (i & 255) * 4;
  int r0 = rowOf[2 * t], r1 = rowOf[2 * t + 1];
  float w0 = tokW[2 * t], w1 = tokW[2 * t + 1];
  float4 a = *(const float4*)(Y + (size_t)r0 * HID + c);
  float4 b = *(const float4*)(Y + (size_t)r1 * HID + c);
  float4 o;
  o.x = w0 * a.x + w1 * b.x; o.y = w0 * a.y + w1 * b.y;
  o.z = w0 * a.z + w1 * b.z; o.w = w0 * a.w + w1 * b.w;
  *(float4*)(out + (size_t)t * HID + c) = o;
}

extern "C" void kernel_launch(void* const* d_in, const int* in_sizes, int n_in,
                              void* d_out, int out_size, void* d_ws, size_t ws_size,
                              hipStream_t stream) {
  const float* x  = (const float*)d_in[0];
  const float* rw = (const float*)d_in[1];
  const float* w1 = (const float*)d_in[2];
  const float* v1 = (const float*)d_in[3];
  const float* w2 = (const float*)d_in[4];
  float* out = (float*)d_out;

  uint8_t* ws = (uint8_t*)d_ws;
  int*   counts = (int*)(ws + 0);                      // 8 ints
  int*   cursor = (int*)(ws + 64);                     // 8 ints
  int*   offs   = (int*)(ws + 128);                    // 9 ints
  int*   tokE   = (int*)(ws + 256);                    // 8192 ints
  float* tokW   = (float*)(ws + 256 + 32768);          // 8192 f32
  int*   rowTok = (int*)(ws + 256 + 65536);            // 8192 ints
  int*   rowOf  = (int*)(ws + 256 + 98304);            // 8192 ints
  uint16_t* G   = (uint16_t*)(ws + 131328);            // (8192+128) x 4096 bf16
  float*    Y   = (float*)(ws + 131328 + (size_t)(T_TOK * TOPK + 128) * FFN * 2);

  hipMemsetAsync(ws, 0, 256, stream);

  moe_router <<<T_TOK / 8, 256, 0, stream>>>(x, rw, counts, tokE, tokW);
  moe_scan   <<<1, 32, 0, stream>>>(counts, offs);
  moe_assign <<<T_TOK / 256, 256, 0, stream>>>(tokE, offs, cursor, rowTok, rowOf);

  dim3 g1(FFN / 64, T_TOK / 128, NEXP);
  moe_glu  <<<g1, 256, 0, stream>>>(x, w1, v1, offs, rowTok, G);
  dim3 g2(HID / 64, T_TOK / 128, NEXP);
  moe_down <<<g2, 256, 0, stream>>>(G, w2, offs, Y);

  moe_combine <<<(T_TOK * HID / 4) / 256, 256, 0, stream>>>(Y, rowOf, tokW, out);
}
